// CrossAttention_18219251269639
// MI455X (gfx1250) — matrix operand order repaired and checked by reading, not compiled, not run on hardware
//
#include <hip/hip_runtime.h>
#include <math.h>

#define BB 8
#define TQN 256
#define TVN 512
#define DD 128
#define LN_EPS 1e-3f

#define QT 16          // queries per workgroup (WMMA M)
#define JCH 64         // context rows per LDS chunk in score phase
#define XP 132         // padded row stride (floats) for D=128 tiles -> conflict-free
#define SCP 520        // padded row stride (floats) for TV=512 score rows

typedef __attribute__((ext_vector_type(16))) __bf16 v16bf;
typedef __attribute__((ext_vector_type(8)))  float  v8f;

__device__ __forceinline__ float fast_tanh(float v) {
#if __has_builtin(__builtin_amdgcn_tanhf)
  return __builtin_amdgcn_tanhf(v);        // v_tanh_f32 (TRANS, co-executes with VALU)
#else
  float r;
  asm("v_tanh_f32 %0, %1" : "=v"(r) : "v"(v));
  return r;
#endif
}

__device__ __forceinline__ float fast_rcp(float v) {
#if __has_builtin(__builtin_amdgcn_rcpf)
  return __builtin_amdgcn_rcpf(v);         // v_rcp_f32 (softmax norm: no IEEE div chain)
#else
  return 1.0f / v;
#endif
}

__device__ __forceinline__ float fast_rsq(float v) {
#if __has_builtin(__builtin_amdgcn_rsqf)
  return __builtin_amdgcn_rsqf(v);         // v_rsq_f32
#else
  return rsqrtf(v);
#endif
}

// A-matrix (16x32 bf16) element->K mapping per ISA 7.12.2:
// VGPR g holds K = {2g, 2g+1} (+8 for lanes 16-31), VGPRs 4-7 are +16.
__device__ __forceinline__ int a_k(int e, int half) {
  int base = (e < 8) ? 0 : 16;
  return base + ((e >> 1) & 3) * 2 + (e & 1) + 8 * half;
}

__global__ __launch_bounds__(256) void addattn_ln_kernel(
    const float* __restrict__ x, const float* __restrict__ ctx,
    const float* __restrict__ scale, const float* __restrict__ gamma,
    const float* __restrict__ beta, float* __restrict__ out) {

  __shared__ float xq[QT * XP];      // x tile, f32
  __shared__ float cf[JCH * XP];     // ctx chunk, f32 (score phase)
  __shared__ float sc[QT * SCP];     // scores -> unnormalized softmax weights
  __shared__ float sl[XP];           // scale vector
  __shared__ float yl[QT * XP];      // attn output tile
  __shared__ float red[QT * 40];     // reductions
  __shared__ float rowmax[QT], rinv[QT], mul[QT], rvl[QT];

  const int t  = threadIdx.x;
  const int b  = blockIdx.y;
  const int q0 = blockIdx.x * QT;

  // ---------- phase 0: load x tile (16x128) and scale, float4 coalesced ----------
  {
    const float4* xg = (const float4*)(x + ((size_t)b * TQN + q0) * DD);
    #pragma unroll
    for (int k = 0; k < 2; ++k) {
      int i = t + 256 * k;                 // 512 float4 total
      int row = i >> 5, c4 = i & 31;
      *(float4*)&xq[row * XP + c4 * 4] = xg[row * (DD / 4) + c4];
    }
    if (t < DD) sl[t] = scale[t];
  }
  __syncthreads();

  const int q = t & 15;    // query row this thread owns in score/softmax phases
  const int g = t >> 4;    // 0..15 group id

  // ---------- phase 1: scores = sum_d scale[d]*tanh(x+ctx), chunked over j ----------
  // Register-block 4 context rows per thread: x/scale LDS reads amortized 4x,
  // 4 independent tanh/fma chains for TRANS<->VALU co-execution.
  for (int jc = 0; jc < TVN; jc += JCH) {
    const float4* cg = (const float4*)(ctx + ((size_t)b * TVN + jc) * DD);
    #pragma unroll
    for (int k = 0; k < 8; ++k) {          // 64*128 floats = 2048 float4
      int i = t + 256 * k;
      int row = i >> 5, c4 = i & 31;
      *(float4*)&cf[row * XP + c4 * 4] = cg[row * (DD / 4) + c4];
    }
    __syncthreads();

    const float* xr  = &xq[q * XP];
    const float* cr0 = &cf[(g     ) * XP];
    const float* cr1 = &cf[(g + 16) * XP];
    const float* cr2 = &cf[(g + 32) * XP];
    const float* cr3 = &cf[(g + 48) * XP];
    float a0 = 0.f, a1 = 0.f, a2 = 0.f, a3 = 0.f;
    #pragma unroll 2
    for (int d = 0; d < DD; d += 4) {
      float4 xv = *(const float4*)(xr  + d);
      float4 sv = *(const float4*)(sl  + d);
      float4 c0 = *(const float4*)(cr0 + d);
      float4 c1 = *(const float4*)(cr1 + d);
      float4 c2 = *(const float4*)(cr2 + d);
      float4 c3 = *(const float4*)(cr3 + d);
      a0 = fmaf(sv.x, fast_tanh(xv.x + c0.x), a0);
      a1 = fmaf(sv.x, fast_tanh(xv.x + c1.x), a1);
      a2 = fmaf(sv.x, fast_tanh(xv.x + c2.x), a2);
      a3 = fmaf(sv.x, fast_tanh(xv.x + c3.x), a3);
      a0 = fmaf(sv.y, fast_tanh(xv.y + c0.y), a0);
      a1 = fmaf(sv.y, fast_tanh(xv.y + c1.y), a1);
      a2 = fmaf(sv.y, fast_tanh(xv.y + c2.y), a2);
      a3 = fmaf(sv.y, fast_tanh(xv.y + c3.y), a3);
      a0 = fmaf(sv.z, fast_tanh(xv.z + c0.z), a0);
      a1 = fmaf(sv.z, fast_tanh(xv.z + c1.z), a1);
      a2 = fmaf(sv.z, fast_tanh(xv.z + c2.z), a2);
      a3 = fmaf(sv.z, fast_tanh(xv.z + c3.z), a3);
      a0 = fmaf(sv.w, fast_tanh(xv.w + c0.w), a0);
      a1 = fmaf(sv.w, fast_tanh(xv.w + c1.w), a1);
      a2 = fmaf(sv.w, fast_tanh(xv.w + c2.w), a2);
      a3 = fmaf(sv.w, fast_tanh(xv.w + c3.w), a3);
    }
    sc[q * SCP + jc + g     ] = a0;
    sc[q * SCP + jc + g + 16] = a1;
    sc[q * SCP + jc + g + 32] = a2;
    sc[q * SCP + jc + g + 48] = a3;
    __syncthreads();
  }

  // ---------- phase 2: softmax over j (per query row), 16 threads per row ----------
  {
    float m = -1e30f;
    for (int k = 0; k < 32; ++k) m = fmaxf(m, sc[q * SCP + g * 32 + k]);
    red[g * 16 + q] = m;
  }
  __syncthreads();
  if (t < QT) {
    float m = red[t];
    for (int gg = 1; gg < 16; ++gg) m = fmaxf(m, red[gg * 16 + t]);
    rowmax[t] = m;
  }
  __syncthreads();
  {
    float m = rowmax[q];
    float s = 0.f;
    for (int k = 0; k < 32; ++k) {
      float e = __expf(sc[q * SCP + g * 32 + k] - m);
      sc[q * SCP + g * 32 + k] = e;        // unnormalized weight
      s += e;
    }
    red[g * 16 + q] = s;
  }
  __syncthreads();
  if (t < QT) {
    float s = 0.f;
    for (int gg = 0; gg < 16; ++gg) s += red[gg * 16 + t];
    rinv[t] = fast_rcp(s);                 // one v_rcp_f32 per row, folded into epilogue
  }
  __syncthreads();

  // ---------- phase 3: attn = W(16x512) @ ctx(512x128) via bf16 WMMA ----------
  {
    const int lane = t & 31;
    const int wave = t >> 5;               // 8 waves -> 8 column tiles of 16
    const int M    = lane & 15;
    const int half = lane >> 4;
    const int n0   = wave * 16;
    const float* cb = ctx + (size_t)b * TVN * DD;

    v8f acc = {0.f, 0.f, 0.f, 0.f, 0.f, 0.f, 0.f, 0.f};
    #pragma unroll 2
    for (int k0 = 0; k0 < TVN; k0 += 32) {
      v16bf av, bv;
      #pragma unroll
      for (int e = 0; e < 16; ++e) {
        av[e] = (__bf16) sc[M * SCP + k0 + a_k(e, half)];
        // B 32x16: same K mapping as A -> K = a_k(e, half)
        bv[e] = (__bf16) cb[(size_t)(k0 + a_k(e, half)) * DD + n0 + M];
      }
      acc = __builtin_amdgcn_wmma_f32_16x16x32_bf16(
          /*neg_a=*/false, av, /*neg_b=*/false, bv,
          /*c_mod=*/(short)0, acc, /*reuse_a=*/false, /*reuse_b=*/false);
    }
    #pragma unroll
    for (int r = 0; r < 8; ++r) {
      int row = r + 8 * half;              // D layout: VGPR r -> M=r / M=r+8
      yl[row * XP + n0 + M] = acc[r] * rinv[row];
    }
  }
  __syncthreads();

  // ---------- phase 4: residual + LayerNorm over D, write out ----------
  {
    const int lq = t >> 4;                 // row
    const int c  = t & 15;                 // 16 threads per row, 8 d's each
    float s1 = 0.f, s2 = 0.f;
    float yv[8];
    #pragma unroll
    for (int p = 0; p < 8; ++p) {
      int d = c + 16 * p;
      float v = yl[lq * XP + d] + xq[lq * XP + d];
      yv[p] = v;
      s1 += v;
      s2 += v * v;
    }
    red[lq * 40 + c]      = s1;
    red[lq * 40 + 20 + c] = s2;
    __syncthreads();
    if (c == 0) {
      float S1 = 0.f, S2 = 0.f;
      for (int k = 0; k < 16; ++k) { S1 += red[lq * 40 + k]; S2 += red[lq * 40 + 20 + k]; }
      float mu  = S1 / (float)DD;
      float var = S2 / (float)DD - mu * mu;
      mul[lq] = mu;
      rvl[lq] = fast_rsq(var + LN_EPS);
    }
    __syncthreads();
    float mu = mul[lq], rv = rvl[lq];
    float* og = out + ((size_t)b * TQN + q0 + lq) * DD;
    #pragma unroll
    for (int p = 0; p < 8; ++p) {
      int d = c + 16 * p;
      og[d] = gamma[d] * (yv[p] - mu) * rv + beta[d];
    }
  }
}

extern "C" void kernel_launch(void* const* d_in, const int* in_sizes, int n_in,
                              void* d_out, int out_size, void* d_ws, size_t ws_size,
                              hipStream_t stream) {
  (void)in_sizes; (void)n_in; (void)out_size; (void)d_ws; (void)ws_size;
  const float* x     = (const float*)d_in[0];
  const float* ctx   = (const float*)d_in[1];
  const float* scale = (const float*)d_in[2];
  const float* gamma = (const float*)d_in[3];
  const float* beta  = (const float*)d_in[4];
  float* out = (float*)d_out;

  dim3 grid(TQN / QT, BB);   // 16 x 8 = 128 workgroups
  addattn_ln_kernel<<<grid, 256, 0, stream>>>(x, ctx, scale, gamma, beta, out);
}
